// DeformableTransformerDecoderLayer_59012850647443
// MI455X (gfx1250) — matrix operand order repaired and checked by reading, not compiled, not run on hardware
//
#include <hip/hip_runtime.h>

// ---------------------------------------------------------------------------
// MI455X (gfx1250) implementation of DeformableTransformerDecoderLayer fwd.
// All dense GEMMs run on v_wmma_f32_16x16x32_f16 (wave32 WMMA, fp32 accum).
// ---------------------------------------------------------------------------

typedef _Float16 half_t;
typedef __attribute__((ext_vector_type(16))) _Float16 v16h;
typedef __attribute__((ext_vector_type(8)))  float    v8f;
typedef __attribute__((ext_vector_type(4)))  unsigned u32x4;

union Frag { v16h v; u32x4 q[2]; };

#define N_    8
#define LQ_   300
#define LIN_  20197
#define RQ_   2400   // N_ * LQ_
#define DM_   256
#define DFFN_ 1024

// ---------------------------- elementwise casts ----------------------------
__global__ __launch_bounds__(256) void k_cast_f16(const float* __restrict__ x,
                                                  half_t* __restrict__ y, long n) {
  long i = (long)blockIdx.x * blockDim.x + threadIdx.x;
  long st = (long)gridDim.x * blockDim.x;
  for (; i < n; i += st) y[i] = (half_t)x[i];
}

__global__ __launch_bounds__(256) void k_addcast_f16(const float* __restrict__ a,
                                                     const float* __restrict__ b,
                                                     half_t* __restrict__ y, long n) {
  long i = (long)blockIdx.x * blockDim.x + threadIdx.x;
  long st = (long)gridDim.x * blockDim.x;
  for (; i < n; i += st) y[i] = (half_t)(a[i] + b[i]);
}

// ------------------------------ WMMA GEMM ----------------------------------
// Y[R,O] = X[R,K] @ W[O,K]^T + bias (+res) (relu?), X/W fp16, accum fp32.
// One wave -> 16 rows x 32 cols (2 accumulators, A reused). Block = 8 waves
// -> 64x64 tile. A/B k order contiguous: k = 16*(lane/16) + s (wave32).
__global__ __launch_bounds__(256) void k_gemm_f16(
    const half_t* __restrict__ X, const half_t* __restrict__ W,
    const float* __restrict__ bias, const float* __restrict__ res,
    float* __restrict__ Yf, half_t* __restrict__ Yh,
    int R, int K, int O, int relu)
{
  const int lane = threadIdx.x & 31;
  const int wid  = threadIdx.x >> 5;
  const int l15  = lane & 15;
  const int hi   = lane >> 4;

  const int r0 = blockIdx.x * 64 + (wid >> 1) * 16;
  const int c0 = blockIdx.y * 64 + (wid & 1) * 32;
  const int c1 = c0 + 16;

  int rowA = r0 + l15; if (rowA >= R) rowA = R - 1;
  // A fragment: lanes 0-15 row=lane K 0..15; lanes 16-31 row=lane-16 K 16..31.
  const half_t* aP  = X + (size_t)rowA * K + hi * 16;
  // B fragment: lanes 0-15 col=lane K 0..15; lanes 16-31 col=lane-16 K 16..31.
  const half_t* bP0 = W + (size_t)(c0 + l15) * K + hi * 16;
  const half_t* bP1 = W + (size_t)(c1 + l15) * K + hi * 16;

  v8f acc0 = {}; v8f acc1 = {};
  for (int k0 = 0; k0 < K; k0 += 32) {
    Frag a, b0, b1;
    a.q[0]  = *(const u32x4*)(aP + k0);
    a.q[1]  = *(const u32x4*)(aP + k0 + 8);
    b0.q[0] = *(const u32x4*)(bP0 + k0);
    b0.q[1] = *(const u32x4*)(bP0 + k0 + 8);
    b1.q[0] = *(const u32x4*)(bP1 + k0);
    b1.q[1] = *(const u32x4*)(bP1 + k0 + 8);
    if (k0 + 32 < K) {              // pull next K-tile toward L2/L0
      __builtin_prefetch(aP + k0 + 32, 0, 0);
      __builtin_prefetch(bP0 + k0 + 32, 0, 0);
      __builtin_prefetch(bP1 + k0 + 32, 0, 0);
    }
    acc0 = __builtin_amdgcn_wmma_f32_16x16x32_f16(false, a.v, false, b0.v,
                                                  (short)0, acc0, false, false);
    acc1 = __builtin_amdgcn_wmma_f32_16x16x32_f16(false, a.v, false, b1.v,
                                                  (short)0, acc1, false, false);
  }

  // C/D layout: VGPR j -> M=j (lanes 0-15), M=j+8 (lanes 16-31); N=lane&15.
  #pragma unroll
  for (int j = 0; j < 8; ++j) {
    int row = r0 + j + hi * 8;
    if (row >= R) continue;
    {
      int col = c0 + l15;
      float v = acc0[j] + bias[col];
      if (res)  v += res[(size_t)row * O + col];
      if (relu) v = v > 0.f ? v : 0.f;
      if (Yf) Yf[(size_t)row * O + col] = v;
      if (Yh) Yh[(size_t)row * O + col] = (half_t)v;
    }
    {
      int col = c1 + l15;
      float v = acc1[j] + bias[col];
      if (res)  v += res[(size_t)row * O + col];
      if (relu) v = v > 0.f ? v : 0.f;
      if (Yf) Yf[(size_t)row * O + col] = v;
      if (Yh) Yh[(size_t)row * O + col] = (half_t)v;
    }
  }
}

// --------------------------- deformable sampling ---------------------------
// One wave per (n, q, head); lane = channel (HEAD_D = 32 = wave32 lanes).
__global__ __launch_bounds__(256) void k_msda_sample(
    const half_t* __restrict__ v,     // [N*LIN, 256]  c = m*32 + d
    const float*  __restrict__ off,   // [N*LQ, 512]   (m,l,p,xy)
    const float*  __restrict__ attn,  // [N*LQ, 128]   (m, l*4+p) pre-softmax
    const float*  __restrict__ ref,   // [N*LQ, 4, 2]
    half_t* __restrict__ samp)        // [N*LQ, 256]
{
  const int wid = threadIdx.x >> 5, lane = threadIdx.x & 31;
  const int w  = blockIdx.x * 8 + wid;    // 0 .. RQ_*8-1
  const int nq = w >> 3;                  // query index (n*LQ+q)
  const int m  = w & 7;                   // head
  const int n  = nq / LQ_;

  // softmax over the 16 (level,point) logits for this head
  const float* ap = attn + (size_t)nq * 128 + m * 16;
  float e[16]; float mx = -1e30f;
  #pragma unroll
  for (int i = 0; i < 16; ++i) { e[i] = ap[i]; mx = fmaxf(mx, e[i]); }
  float s = 0.f;
  #pragma unroll
  for (int i = 0; i < 16; ++i) { e[i] = __expf(e[i] - mx); s += e[i]; }
  const float inv = 1.f / s;

  const float* op = off + (size_t)nq * 512 + m * 32;  // 16 (x,y) pairs
  const float* rp = ref + (size_t)nq * 8;

  const int Hs[4] = {100, 50, 25, 13};
  const int Ws[4] = {152, 76, 38, 19};
  const int St[4] = {0, 15200, 19000, 19950};

  float acc = 0.f;
  #pragma unroll
  for (int l = 0; l < 4; ++l) {
    const int H = Hs[l], Wd = Ws[l], st = St[l];
    const float rx = rp[l * 2 + 0], ry = rp[l * 2 + 1];
    #pragma unroll
    for (int p = 0; p < 4; ++p) {
      const float ox = op[(l * 4 + p) * 2 + 0];
      const float oy = op[(l * 4 + p) * 2 + 1];
      // loc = ref + off/[W,H];  x = loc_x*W - 0.5
      const float x = rx * (float)Wd + ox - 0.5f;
      const float y = ry * (float)H  + oy - 0.5f;
      const float x0f = floorf(x), y0f = floorf(y);
      const float lx = x - x0f, ly = y - y0f;
      const int x0 = (int)x0f, y0 = (int)y0f;
      const float wp = e[l * 4 + p] * inv;
      const float cw[4] = {(1.f-lx)*(1.f-ly), lx*(1.f-ly), (1.f-lx)*ly, lx*ly};
      const int xs[4] = {x0, x0 + 1, x0, x0 + 1};
      const int ys[4] = {y0, y0, y0 + 1, y0 + 1};
      #pragma unroll
      for (int c = 0; c < 4; ++c) {
        const int xi = xs[c], yi = ys[c];
        if (xi >= 0 && xi < Wd && yi >= 0 && yi < H) {
          const int idx = st + yi * Wd + xi;
          const float vv = (float)v[((size_t)n * LIN_ + idx) * 256 + m * 32 + lane];
          acc += wp * cw[c] * vv;
        }
      }
    }
  }
  samp[(size_t)nq * 256 + m * 32 + lane] = (half_t)acc;
}

// ------------------------------- layer norm --------------------------------
__global__ __launch_bounds__(256) void k_layernorm256(
    const float* __restrict__ x, const float* __restrict__ g,
    const float* __restrict__ b, float* __restrict__ yf,
    half_t* __restrict__ yh, int rows)
{
  const int lane = threadIdx.x & 31, wid = threadIdx.x >> 5;
  const int row = blockIdx.x * 8 + wid;
  if (row >= rows) return;
  const float* xr = x + (size_t)row * 256;
  float v[8]; float s = 0.f, ss = 0.f;
  #pragma unroll
  for (int i = 0; i < 8; ++i) {
    v[i] = xr[lane + 32 * i]; s += v[i]; ss += v[i] * v[i];
  }
  #pragma unroll
  for (int m = 16; m > 0; m >>= 1) {            // wave32 tree reduce
    s  += __shfl_xor(s,  m, 32);
    ss += __shfl_xor(ss, m, 32);
  }
  const float mu   = s * (1.f / 256.f);
  const float var  = ss * (1.f / 256.f) - mu * mu;
  const float rstd = rsqrtf(var + 1e-5f);
  #pragma unroll
  for (int i = 0; i < 8; ++i) {
    const int c = lane + 32 * i;
    const float o = (v[i] - mu) * rstd * g[c] + b[c];
    if (yf) yf[(size_t)row * 256 + c] = o;
    if (yh) yh[(size_t)row * 256 + c] = (half_t)o;
  }
}

// -------------------------------- launcher ---------------------------------
static inline size_t alignUp256(size_t x) { return (x + 255) & ~(size_t)255; }

extern "C" void kernel_launch(void* const* d_in, const int* in_sizes, int n_in,
                              void* d_out, int out_size, void* d_ws, size_t ws_size,
                              hipStream_t stream)
{
  (void)in_sizes; (void)n_in; (void)out_size; (void)ws_size;
  const long RV = (long)N_ * LIN_;   // 161576 value rows
  const int  RQ = RQ_;

  const float* in_tgt      = (const float*)d_in[0];
  const float* in_pre_tgt  = (const float*)d_in[1];
  const float* in_qpos     = (const float*)d_in[2];
  const float* in_pre_qpos = (const float*)d_in[3];
  const float* in_ref      = (const float*)d_in[4];
  const float* in_src      = (const float*)d_in[5];
  const float* in_pre_src  = (const float*)d_in[10];
  const float* in_pre_ref  = (const float*)d_in[11];
  const float* cw[8]; for (int i = 0; i < 8; ++i) cw[i]  = (const float*)d_in[12 + i];
  const float* sw[8]; for (int i = 0; i < 8; ++i) sw[i]  = (const float*)d_in[20 + i];
  const float* lng[8]; for (int i = 0; i < 8; ++i) lng[i] = (const float*)d_in[28 + i];
  const float* ffw[8]; for (int i = 0; i < 8; ++i) ffw[i] = (const float*)d_in[36 + i];

  // ---- workspace carve ----
  size_t woff = 0; char* base = (char*)d_ws;
  auto carve = [&](size_t bytes) -> char* {
    char* p = base + woff; woff += alignUp256(bytes); return p;
  };
  half_t* h_vw_c = (half_t*)carve((size_t)DM_ * DM_ * 2);
  half_t* h_ow_c = (half_t*)carve((size_t)512 * DM_ * 2);
  half_t* h_aw_c = (half_t*)carve((size_t)128 * DM_ * 2);
  half_t* h_pw_c = (half_t*)carve((size_t)DM_ * DM_ * 2);
  half_t* h_vw_s = (half_t*)carve((size_t)DM_ * DM_ * 2);
  half_t* h_ow_s = (half_t*)carve((size_t)512 * DM_ * 2);
  half_t* h_aw_s = (half_t*)carve((size_t)128 * DM_ * 2);
  half_t* h_pw_s = (half_t*)carve((size_t)DM_ * DM_ * 2);
  half_t* h_l1   = (half_t*)carve((size_t)DFFN_ * DM_ * 2);
  half_t* h_l2   = (half_t*)carve((size_t)DM_ * DFFN_ * 2);
  half_t* h_pl1  = (half_t*)carve((size_t)DFFN_ * DM_ * 2);
  half_t* h_pl2  = (half_t*)carve((size_t)DM_ * DFFN_ * 2);
  half_t* x_h    = (half_t*)carve((size_t)RV * DM_ * 2);   // src fp16 (reused)
  half_t* v_h    = (half_t*)carve((size_t)RV * DM_ * 2);   // value fp16 (reused)
  half_t* q_h    = (half_t*)carve((size_t)RQ * DM_ * 2);
  float*  off_f  = (float*) carve((size_t)RQ * 512 * 4);
  float*  attn_f = (float*) carve((size_t)RQ * 128 * 4);
  half_t* samp_h = (half_t*)carve((size_t)RQ * DM_ * 2);
  float*  tgt1   = (float*) carve((size_t)RQ * DM_ * 4);
  float*  ptgt1  = (float*) carve((size_t)RQ * DM_ * 4);
  float*  t_f    = (float*) carve((size_t)RQ * DM_ * 4);
  half_t* t_h    = (half_t*)carve((size_t)RQ * DM_ * 2);
  half_t* hid_h  = (half_t*)carve((size_t)RQ * DFFN_ * 2);
  float*  y_f    = (float*) carve((size_t)RQ * DM_ * 4);

  auto castK = [&](const float* xf, half_t* yh, long n) {
    long blk = (n + 255) / 256; if (blk > 8192) blk = 8192;
    k_cast_f16<<<dim3((unsigned)blk), 256, 0, stream>>>(xf, yh, n);
  };
  auto gemm = [&](const half_t* X, const half_t* W, const float* bias,
                  const float* res, float* Yf, half_t* Yh,
                  int R, int K, int O, int relu) {
    dim3 grid((unsigned)((R + 63) / 64), (unsigned)(O / 64));
    k_gemm_f16<<<grid, 256, 0, stream>>>(X, W, bias, res, Yf, Yh, R, K, O, relu);
  };

  // ---- one-time fp16 weight conversion ----
  castK(cw[0], h_vw_c, DM_ * DM_);  castK(cw[2], h_ow_c, 512 * DM_);
  castK(cw[4], h_aw_c, 128 * DM_);  castK(cw[6], h_pw_c, DM_ * DM_);
  castK(sw[0], h_vw_s, DM_ * DM_);  castK(sw[2], h_ow_s, 512 * DM_);
  castK(sw[4], h_aw_s, 128 * DM_);  castK(sw[6], h_pw_s, DM_ * DM_);
  castK(ffw[0], h_l1, DFFN_ * DM_); castK(ffw[2], h_l2, DM_ * DFFN_);
  castK(ffw[4], h_pl1, DFFN_ * DM_); castK(ffw[6], h_pl2, DM_ * DFFN_);

  auto run_msda = [&](const float* srcp, const float* basep, const float* posp,
                      const float* refp,
                      half_t* vw, half_t* ow, half_t* aw, half_t* pw,
                      const float* vb, const float* ob, const float* ab,
                      const float* pb, float* outp) {
    castK(srcp, x_h, RV * DM_);
    gemm(x_h, vw, vb, nullptr, nullptr, v_h, (int)RV, DM_, DM_, 0);   // value proj
    k_addcast_f16<<<2400, 256, 0, stream>>>(basep, posp, q_h, (long)RQ * DM_);
    gemm(q_h, ow, ob, nullptr, off_f, nullptr, RQ, DM_, 512, 0);      // offsets
    gemm(q_h, aw, ab, nullptr, attn_f, nullptr, RQ, DM_, 128, 0);     // attn logits
    k_msda_sample<<<RQ, 256, 0, stream>>>(v_h, off_f, attn_f, refp, samp_h);
    gemm(samp_h, pw, pb, basep, outp, nullptr, RQ, DM_, DM_, 0);      // out proj + residual
  };

  // self attention branch (pre_*), then cross branch (reuses x_h/v_h)
  run_msda(in_pre_src, in_pre_tgt, in_pre_qpos, in_pre_ref,
           h_vw_s, h_ow_s, h_aw_s, h_pw_s, sw[1], sw[3], sw[5], sw[7], ptgt1);
  run_msda(in_src, in_tgt, in_qpos, in_ref,
           h_vw_c, h_ow_c, h_aw_c, h_pw_c, cw[1], cw[3], cw[5], cw[7], tgt1);

  float* out_t  = (float*)d_out;
  float* out_pt = out_t + (size_t)RQ * DM_;

  auto run_ffn = [&](const float* x1, const float* g1, const float* b1,
                     half_t* w1, const float* bias1, half_t* w2,
                     const float* bias2, const float* g3, const float* b3,
                     float* outp) {
    k_layernorm256<<<300, 256, 0, stream>>>(x1, g1, b1, t_f, t_h, RQ);
    gemm(t_h, w1, bias1, nullptr, nullptr, hid_h, RQ, DM_, DFFN_, 1); // lin1+relu
    gemm(hid_h, w2, bias2, t_f, y_f, nullptr, RQ, DFFN_, DM_, 0);     // lin2+residual
    k_layernorm256<<<300, 256, 0, stream>>>(y_f, g3, b3, outp, nullptr, RQ);
  };

  run_ffn(tgt1,  lng[0], lng[1], h_l1,  ffw[1], h_l2,  ffw[3], lng[4], lng[5], out_t);
  run_ffn(ptgt1, lng[2], lng[3], h_pl1, ffw[5], h_pl2, ffw[7], lng[6], lng[7], out_pt);
}
